// PostDecoderSATransformerBlock_5815385719388
// MI455X (gfx1250) — matrix operand order repaired and checked by reading, not compiled, not run on hardware
//
#include <hip/hip_runtime.h>
#include <math.h>

// Problem constants (match reference)
#define BB 4
#define SS 1024
#define DD 2048
#define HH 16
#define HDIM 128
#define II 8192
#define MT (BB*SS)          // 4096 token rows

typedef __bf16 bf16;
typedef __attribute__((ext_vector_type(16))) __bf16 v16bf;
typedef __attribute__((ext_vector_type(8)))  __bf16 v8bf;
typedef __attribute__((ext_vector_type(4)))  __bf16 v4bf;
typedef __attribute__((ext_vector_type(8)))  float  v8f;

__device__ __forceinline__ v16bf cat8(v8bf a, v8bf b) {
  return __builtin_shufflevector(a, b, 0,1,2,3,4,5,6,7,8,9,10,11,12,13,14,15);
}

__device__ __forceinline__ v8f wmma_bf16(v16bf a, v16bf b, v8f c) {
  // D(f32 16x16) = A(16x32 bf16) x B(32x16 bf16) + C
  return __builtin_amdgcn_wmma_f32_16x16x32_bf16(false, a, false, b, (short)0, c, false, false);
}

// Async DMA: global (16B per lane) -> LDS, tracked by ASYNCcnt
__device__ __forceinline__ void async_copy_b128(const void* gaddr, uint32_t lds_addr) {
  asm volatile("global_load_async_to_lds_b128 %0, %1, off"
               :: "v"(lds_addr), "v"((unsigned long long)(uintptr_t)gaddr) : "memory");
}
__device__ __forceinline__ void wait_async0() {
  asm volatile("s_wait_asynccnt 0" ::: "memory");
}

// ---------------------------------------------------------------- fp32 -> bf16
__global__ void cvt_bf16_kernel(const float* __restrict__ src, bf16* __restrict__ dst, int n) {
  int i = blockIdx.x * blockDim.x + threadIdx.x;
  int stride = gridDim.x * blockDim.x;
  for (; i < n; i += stride) dst[i] = (bf16)src[i];
}

// ---------------------------------------------------------------- LayerNorm (f32 in, bf16 out)
__global__ void ln_bf16_kernel(const float* __restrict__ x, const float* __restrict__ w,
                               const float* __restrict__ b, bf16* __restrict__ out) {
  __shared__ float s1[256], s2[256];
  const int row = blockIdx.x, tid = threadIdx.x;
  const float* xr = x + (size_t)row * DD;
  float sum = 0.f, sq = 0.f;
  for (int c = tid; c < DD; c += 256) { float v = xr[c]; sum += v; sq += v * v; }
  s1[tid] = sum; s2[tid] = sq; __syncthreads();
  for (int off = 128; off > 0; off >>= 1) {
    if (tid < off) { s1[tid] += s1[tid + off]; s2[tid] += s2[tid + off]; }
    __syncthreads();
  }
  const float mu = s1[0] * (1.0f / DD);
  const float var = s2[0] * (1.0f / DD) - mu * mu;   // biased variance (jnp.var)
  const float rs = rsqrtf(var + 1e-5f);
  for (int c = tid; c < DD; c += 256)
    out[(size_t)row * DD + c] = (bf16)((xr[c] - mu) * rs * w[c] + b[c]);
}

// ---------------------------------------------------------------- WMMA GEMM, LDS double-buffered via async DMA
// out = epilogue(A[M,K] @ W[N,K]^T + bias)
// mode 0: bf16 store (scaled)   mode 1: exact-GELU -> bf16   mode 2: f32 = resid + val
// Block tile 128M x 128N, 8 waves (2Mx4N), wave tile 64M x 32N, k-chunk 32.
#define GPAD 40   // LDS row pitch (bf16): 80B -> 16B-aligned reads, 20n mod 64 banks (conflict-free)
__global__ __launch_bounds__(256)
void wmma_gemm_kernel(const bf16* __restrict__ A, const bf16* __restrict__ W,
                      const float* __restrict__ bias, const float* __restrict__ resid,
                      bf16* __restrict__ outb, float* __restrict__ outf,
                      int M, int N, int K, float scale, int mode) {
  __shared__ bf16 Alds[2][128][GPAD];
  __shared__ bf16 Blds[2][128][GPAD];

  const int tid  = threadIdx.x;
  const int lane = tid & 31;
  const int wid  = tid >> 5;
  const int hl   = lane >> 4;          // lane half selects K sub-ranges per ISA layout
  const int l16  = lane & 15;
  const int wm   = wid >> 2;           // 0..1  (M direction)
  const int wn   = wid & 3;            // 0..3  (N direction)
  const int mbase = blockIdx.x * 128;
  const int nbase = blockIdx.y * 128;

  v8f acc[4][2] = {};
  const int nk = K / 32;

  // Stage chunk j (A 128x32 + B 128x32 = 1024 x 16B transfers; 4 per thread) into buffer buf.
  auto stage = [&](int j, int buf) {
#pragma unroll
    for (int i = 0; i < 2; i++) {
      const int idx = tid * 2 + i;       // 0..511
      const int row = idx >> 2;          // 0..127
      const int seg = idx & 3;           // 16B segment within 64B k-chunk row
      const int kof = j * 32 + seg * 8;
      async_copy_b128(A + (size_t)(mbase + row) * K + kof,
                      (uint32_t)(uintptr_t)&Alds[buf][row][seg * 8]);
      async_copy_b128(W + (size_t)(nbase + row) * K + kof,
                      (uint32_t)(uintptr_t)&Blds[buf][row][seg * 8]);
    }
  };

  stage(0, 0);
  for (int j = 0; j < nk; j++) {
    const int buf = j & 1;
    wait_async0();        // my share of chunk j is in LDS
    __syncthreads();      // everyone's share is in LDS; everyone done reading buf^1
    if (j + 1 < nk) stage(j + 1, buf ^ 1);   // DMA next chunk under the compute below

    v16bf bfr[2];
#pragma unroll
    for (int t = 0; t < 2; t++)
      bfr[t] = cat8(*(const v8bf*)&Blds[buf][wn * 32 + t * 16 + l16][hl * 8], *(const v8bf*)&Blds[buf][wn * 32 + t * 16 + l16][16 + hl * 8]);
#pragma unroll
    for (int mt = 0; mt < 4; mt++) {
      const bf16* ar = &Alds[buf][wm * 64 + mt * 16 + l16][0];
      v16bf af = cat8(*(const v8bf*)(ar + hl * 8), *(const v8bf*)(ar + 16 + hl * 8));
#pragma unroll
      for (int t = 0; t < 2; t++)
        acc[mt][t] = wmma_bf16(af, bfr[t], acc[mt][t]);
    }
  }

#pragma unroll
  for (int t = 0; t < 2; t++) {
    const int col = nbase + wn * 32 + t * 16 + l16;
    const float bv = bias[col];
#pragma unroll
    for (int mt = 0; mt < 4; mt++) {
#pragma unroll
      for (int r = 0; r < 8; r++) {
        const int row = mbase + wm * 64 + mt * 16 + r + 8 * hl;   // C layout: VGPR r -> M = r + 8*half
        float v = (acc[mt][t][r] + bv) * scale;
        const size_t idx = (size_t)row * N + col;
        if (mode == 0)      outb[idx] = (bf16)v;
        else if (mode == 1) outb[idx] = (bf16)(0.5f * v * (1.0f + erff(v * 0.70710678118654752f)));
        else                outf[idx] = resid[idx] + v;
      }
    }
  }
}

// ---------------------------------------------------------------- Flash attention (one wave = 16 q-rows of one head)
#define VPAD 40   // row pad (bf16): 80B rows -> 16B-aligned b128 LDS reads, conflict-free (20n mod 64)
__global__ __launch_bounds__(128)
void attn_kernel(const bf16* __restrict__ Q, const bf16* __restrict__ Km,
                 const bf16* __restrict__ V, const float* __restrict__ mask,
                 bf16* __restrict__ O) {
  __shared__ bf16 Plds[4][16][VPAD];      // probabilities tile 16x32 (bf16) per wave
  __shared__ bf16 VT[4][HDIM][VPAD];      // V chunk transposed: [hd][k] per wave

  const int lane = threadIdx.x & 31;
  const int wid  = threadIdx.x >> 5;
  const int hl   = lane >> 4;
  const int l16  = lane & 15;
  const int qt = blockIdx.x * 4 + wid;
  const int h  = blockIdx.y, b = blockIdx.z;
  const int qbase = qt * 16;
  const size_t base = (size_t)b * SS * DD + (size_t)h * HDIM;   // [b, s=0, h, 0]

  // Preload Q fragments for all 4 K-dim (head-dim) chunks
  v16bf qf[4];
  {
    const bf16* qrow = Q + base + (size_t)(qbase + l16) * DD;
#pragma unroll
    for (int ks = 0; ks < 4; ks++) {
      v8bf a0 = *(const v8bf*)(qrow + ks * 32 + hl * 8);
      v8bf a1 = *(const v8bf*)(qrow + ks * 32 + 16 + hl * 8);
      qf[ks] = cat8(a0, a1);
    }
  }

  v8f o[8] = {};
  float mrow[8], lrow[8];
#pragma unroll
  for (int r = 0; r < 8; r++) { mrow[r] = -INFINITY; lrow[r] = 0.f; }

  for (int kb = 0; kb < SS; kb += 32) {
    // additive mask per key column (reference: (1-mask)!=0 -> -FLT_MAX else 0)
    float am0 = 1.0f - mask[(size_t)b * SS + kb + l16];
    float am1 = 1.0f - mask[(size_t)b * SS + kb + 16 + l16];
    const float add0 = (am0 != 0.f) ? -3.40282347e38f : am0;
    const float add1 = (am1 != 0.f) ? -3.40282347e38f : am1;

    // scores: two 16x16 tiles over this 32-key chunk (K-dim = head dim 128)
    v8f s0 = {}, s1 = {};
#pragma unroll
    for (int ks = 0; ks < 4; ks++) {
      const bf16* k0 = Km + base + (size_t)(kb + l16) * DD + ks * 32 + hl * 8;
      const bf16* k1 = Km + base + (size_t)(kb + 16 + l16) * DD + ks * 32 + hl * 8;
      s0 = wmma_bf16(qf[ks], cat8(*(const v8bf*)k0, *(const v8bf*)(k0 + 16)), s0);
      s1 = wmma_bf16(qf[ks], cat8(*(const v8bf*)k1, *(const v8bf*)(k1 + 16)), s1);
    }

    // online softmax update (per-row stats replicated across each 16-lane group)
    float e0[8], e1[8];
#pragma unroll
    for (int r = 0; r < 8; r++) {
      float a = s0[r] + add0, c = s1[r] + add1;
      float t = fmaxf(a, c);
      t = fmaxf(t, __shfl_xor(t, 1, 32));
      t = fmaxf(t, __shfl_xor(t, 2, 32));
      t = fmaxf(t, __shfl_xor(t, 4, 32));
      t = fmaxf(t, __shfl_xor(t, 8, 32));
      const float mn = fmaxf(mrow[r], t);
      float alpha;
      if (mn == -INFINITY) { alpha = 1.f; e0[r] = 0.f; e1[r] = 0.f; }
      else {
        alpha = (mrow[r] == -INFINITY) ? 0.f : __expf(mrow[r] - mn);
        e0[r] = __expf(a - mn);
        e1[r] = __expf(c - mn);
      }
      mrow[r] = mn;
      float cs = e0[r] + e1[r];
      cs += __shfl_xor(cs, 1, 32);
      cs += __shfl_xor(cs, 2, 32);
      cs += __shfl_xor(cs, 4, 32);
      cs += __shfl_xor(cs, 8, 32);
      lrow[r] = lrow[r] * alpha + cs;
#pragma unroll
      for (int t8 = 0; t8 < 8; t8++) o[t8][r] *= alpha;
      // store probabilities as bf16 (C layout -> LDS row-major 16x32)
      Plds[wid][r + 8 * hl][l16]      = (bf16)e0[r];
      Plds[wid][r + 8 * hl][16 + l16] = (bf16)e1[r];
    }

    // stage V chunk [32 x 128] into LDS transposed [hd][k]
    const bf16* vb = V + base + (size_t)kb * DD;
#pragma unroll 4
    for (int rr = 0; rr < 32; rr++) {
      v4bf vv = *(const v4bf*)(vb + (size_t)rr * DD + lane * 4);
#pragma unroll
      for (int i = 0; i < 4; i++) VT[wid][lane * 4 + i][rr] = vv[i];
    }
    asm volatile("s_wait_dscnt 0" ::: "memory");   // LDS visible before cross-lane re-read

    // P as A-fragment (exact ISA 16-bit A layout), V columns as B-fragments
    v8bf pa0 = *(const v8bf*)&Plds[wid][l16][hl * 8];
    v8bf pa1 = *(const v8bf*)&Plds[wid][l16][16 + hl * 8];
    v16bf pf = cat8(pa0, pa1);
#pragma unroll
    for (int t8 = 0; t8 < 8; t8++) {
      v16bf vf = cat8(*(const v8bf*)&VT[wid][t8 * 16 + l16][hl * 8], *(const v8bf*)&VT[wid][t8 * 16 + l16][16 + hl * 8]);
      o[t8] = wmma_bf16(pf, vf, o[t8]);
    }
  }

  // normalize and write attention output (bf16, [B,S,D] with head offset)
#pragma unroll
  for (int r = 0; r < 8; r++) {
    const float inv = (lrow[r] > 0.f) ? 1.f / lrow[r] : 0.f;
    const size_t rowoff = base + (size_t)(qbase + r + 8 * hl) * DD;
#pragma unroll
    for (int t8 = 0; t8 < 8; t8++)
      O[rowoff + t8 * 16 + l16] = (bf16)(o[t8][r] * inv);
  }
}

// ---------------------------------------------------------------- host
extern "C" void kernel_launch(void* const* d_in, const int* in_sizes, int n_in,
                              void* d_out, int out_size, void* d_ws, size_t ws_size,
                              hipStream_t stream) {
  (void)in_sizes; (void)n_in; (void)out_size; (void)ws_size;
  const float* hidden_in = (const float*)d_in[1];
  const float* amask     = (const float*)d_in[2];
  const float* Wq = (const float*)d_in[4];  const float* bq = (const float*)d_in[5];
  const float* Wk = (const float*)d_in[6];  const float* bk = (const float*)d_in[7];
  const float* Wv = (const float*)d_in[8];  const float* bv = (const float*)d_in[9];
  const float* Wo = (const float*)d_in[10]; const float* bo = (const float*)d_in[11];
  const float* ln1w = (const float*)d_in[12]; const float* ln1b = (const float*)d_in[13];
  const float* ln2w = (const float*)d_in[14]; const float* ln2b = (const float*)d_in[15];
  const float* W1 = (const float*)d_in[16]; const float* b1 = (const float*)d_in[17];
  const float* W2 = (const float*)d_in[18]; const float* b2 = (const float*)d_in[19];
  float* out = (float*)d_out;

  char* ws = (char*)d_ws;
  size_t off = 0;
  auto alloc = [&](size_t bytes) { void* p = ws + off; off += (bytes + 255) & ~(size_t)255; return p; };
  bf16* Wq_b = (bf16*)alloc((size_t)DD * DD * 2);
  bf16* Wk_b = (bf16*)alloc((size_t)DD * DD * 2);
  bf16* Wv_b = (bf16*)alloc((size_t)DD * DD * 2);
  bf16* Wo_b = (bf16*)alloc((size_t)DD * DD * 2);
  bf16* W1_b = (bf16*)alloc((size_t)II * DD * 2);
  bf16* W2_b = (bf16*)alloc((size_t)DD * II * 2);
  bf16* x_b   = (bf16*)alloc((size_t)MT * DD * 2);
  bf16* q_b   = (bf16*)alloc((size_t)MT * DD * 2);
  bf16* k_b   = (bf16*)alloc((size_t)MT * DD * 2);
  bf16* v_b   = (bf16*)alloc((size_t)MT * DD * 2);
  bf16* at_b  = (bf16*)alloc((size_t)MT * DD * 2);
  float* hid  = (float*)alloc((size_t)MT * DD * 4);
  bf16* y_b   = (bf16*)alloc((size_t)MT * DD * 2);
  bf16* h1_b  = (bf16*)alloc((size_t)MT * II * 2);

  // 1) weights -> bf16
  cvt_bf16_kernel<<<2048, 256, 0, stream>>>(Wq, Wq_b, DD * DD);
  cvt_bf16_kernel<<<2048, 256, 0, stream>>>(Wk, Wk_b, DD * DD);
  cvt_bf16_kernel<<<2048, 256, 0, stream>>>(Wv, Wv_b, DD * DD);
  cvt_bf16_kernel<<<2048, 256, 0, stream>>>(Wo, Wo_b, DD * DD);
  cvt_bf16_kernel<<<4096, 256, 0, stream>>>(W1, W1_b, II * DD);
  cvt_bf16_kernel<<<4096, 256, 0, stream>>>(W2, W2_b, DD * II);

  // 2) LN1
  ln_bf16_kernel<<<MT, 256, 0, stream>>>(hidden_in, ln1w, ln1b, x_b);

  // 3) Q/K/V projections
  const float sc = 0.08838834764831845f;   // HD^-0.5
  dim3 gD(MT / 128, DD / 128);
  wmma_gemm_kernel<<<gD, 256, 0, stream>>>(x_b, Wq_b, bq, nullptr, q_b, nullptr, MT, DD, DD, sc,  0);
  wmma_gemm_kernel<<<gD, 256, 0, stream>>>(x_b, Wk_b, bk, nullptr, k_b, nullptr, MT, DD, DD, 1.f, 0);
  wmma_gemm_kernel<<<gD, 256, 0, stream>>>(x_b, Wv_b, bv, nullptr, v_b, nullptr, MT, DD, DD, 1.f, 0);

  // 4) attention
  attn_kernel<<<dim3(SS / 16 / 4, HH, BB), 128, 0, stream>>>(q_b, k_b, v_b, amask, at_b);

  // 5) O-projection + residual -> hidden (f32)
  wmma_gemm_kernel<<<gD, 256, 0, stream>>>(at_b, Wo_b, bo, hidden_in, nullptr, hid, MT, DD, DD, 1.f, 2);

  // 6) LN2, MLP (GELU), final residual
  ln_bf16_kernel<<<MT, 256, 0, stream>>>(hid, ln2w, ln2b, y_b);
  wmma_gemm_kernel<<<dim3(MT / 128, II / 128), 256, 0, stream>>>(y_b, W1_b, b1, nullptr, h1_b, nullptr, MT, II, DD, 1.f, 1);
  wmma_gemm_kernel<<<gD, 256, 0, stream>>>(h1_b, W2_b, b2, hid, nullptr, out, MT, DD, II, 1.f, 2);
}
